// RWSEEncoder_76879914599071
// MI455X (gfx1250) — compile-verified, matrix operand order checked by reading, not run on hardware
//
#include <hip/hip_runtime.h>
#include <hip/hip_bf16.h>

#define WALK   16
#define ODIM   8
#define CCHUNK 500
#define TPB    256

typedef __attribute__((ext_vector_type(2)))  float    v2f;
typedef __attribute__((ext_vector_type(8)))  float    v8f;
typedef __attribute__((ext_vector_type(16))) _Float16 v16h;

// ---------------- CSC construction (deterministic) ----------------

__global__ void rwse_zero_i32(int* p, int n) {
    int i = blockIdx.x * blockDim.x + threadIdx.x;
    if (i < n) p[i] = 0;
}

__global__ void rwse_count_in(const int* __restrict__ col, int E, int* __restrict__ cnt) {
    int e = blockIdx.x * blockDim.x + threadIdx.x;
    if (e < E) atomicAdd(&cnt[col[e]], 1);   // integer atomics: deterministic counts
}

__global__ void rwse_scan_off(const int* __restrict__ cnt, int* __restrict__ off, int N) {
    if (blockIdx.x == 0 && threadIdx.x == 0) {
        int acc = 0;
        for (int i = 0; i < N; ++i) { off[i] = acc; acc += cnt[i]; }
        off[N] = acc;
    }
}

// Stable, order-deterministic CSC fill: one thread per destination column scans
// the edge list in order. col[e] load is wave-uniform -> broadcast from L2.
__global__ void rwse_fill_csc(const int* __restrict__ row, const int* __restrict__ col,
                              const float* __restrict__ val, int E,
                              const int* __restrict__ off,
                              int* __restrict__ rows_s, float* __restrict__ vals_s, int N) {
    int c = blockIdx.x * blockDim.x + threadIdx.x;
    if (c >= N) return;
    int p = off[c];
    for (int e = 0; e < E; ++e) {
        if (col[e] == c) { rows_s[p] = row[e]; vals_s[p] = val[e]; ++p; }
    }
}

// ---------------- chunked identity-block power iteration ----------------

__global__ void rwse_init_onehot(float* __restrict__ cur, int start) {
    int i = blockIdx.x;            // node (row of cur)
    int t = threadIdx.x;           // column owner: t and t+TPB
    float* r = cur + (size_t)i * CCHUNK;
    r[t] = (i == start + t) ? 1.f : 0.f;
    if (t + TPB < CCHUNK) r[t + TPB] = (i == start + t + TPB) ? 1.f : 0.f;
}

// nxt[c, :] = sum_{e: col[e]==c} rw[e] * cur[row[e], :]   (pull, register accum)
// Also extracts this step's diagonal entries for nodes inside the chunk.
__global__ void rwse_spmm_step(const int* __restrict__ off, const int* __restrict__ rows_s,
                               const float* __restrict__ vals_s,
                               const float* __restrict__ cur, float* __restrict__ nxt,
                               float* __restrict__ diag, int start, int k) {
    int c   = blockIdx.x;          // destination node
    int t   = threadIdx.x;
    int beg = off[c], end = off[c + 1];
    bool has2 = (t + TPB) < CCHUNK;
    float a0 = 0.f, a1 = 0.f;
    for (int e = beg; e < end; ++e) {
        float v = vals_s[e];
        const float* cr = cur + (size_t)rows_s[e] * CCHUNK;
        a0 += v * cr[t];
        if (has2) a1 += v * cr[t + TPB];
    }
    float* nr = nxt + (size_t)c * CCHUNK;
    nr[t] = a0;
    if (has2) nr[t + TPB] = a1;

    int d = c - start;             // diagonal column of node c within this chunk
    if (d >= 0 && d < CCHUNK) {
        if (d < TPB)      { if (t == d)        diag[c * WALK + k] = a0; }
        else              { if (t == d - TPB)  diag[c * WALK + k] = a1; }
    }
}

// ---------------- final projection: out = diag @ W^T + b via WMMA ----------------
// One wave32 per 16-node tile. M=16 nodes, N=16 (8 used), K=16.
// Loads are unconditional (index-clamped) and masked by multiply, so no
// exec-mask diamonds appear around the WMMA inputs; pairs load as b64.

__global__ __launch_bounds__(32)
void rwse_linear_wmma(const float* __restrict__ diag, const float* __restrict__ W,
                      const float* __restrict__ b, float* __restrict__ out, int N) {
    int lane = threadIdx.x;        // 0..31 (wave32)
    int m    = lane & 15;          // A-matrix row (node within tile); also B/D column
    int hi   = lane >> 4;          // lane half selects K placement
    int base = blockIdx.x * 16;
    int node = base + m;

    // clamp + mask instead of branching around loads
    int   nodec = (node < N) ? node : 0;
    float amask = (node < N) ? 1.f : 0.f;
    int   n     = m;               // output feature column for B/D
    int   nc    = n & (ODIM - 1);  // clamped W row
    float bmask = (n < ODIM) ? 1.f : 0.f;

    // rows are 16 floats (64B); k0 = 4q + 2*hi is even -> 8B-aligned v2f pairs
    const v2f* dg = (const v2f*)(diag + (size_t)nodec * WALK);
    const v2f* wr = (const v2f*)(W    + (size_t)nc    * WALK);

    v8f acc = {};
#if __has_builtin(__builtin_amdgcn_wmma_f32_16x16x4_f32)
    #pragma unroll
    for (int q = 0; q < 4; ++q) {          // A/B element j holds K = 4q + 2*hi + j
        v2f a  = dg[2 * q + hi];
        v2f bb = wr[2 * q + hi];
        a.x  *= amask; a.y  *= amask;
        bb.x *= bmask; bb.y *= bmask;
        acc = __builtin_amdgcn_wmma_f32_16x16x4_f32(false, a, false, bb,
                                                    (short)0, acc, false, false);
    }
#else
    // Fallback: codegen-confirmed f16 WMMA, K=16 padded to 32 with zeros.
    v16h a = {}, bb = {};
    #pragma unroll
    for (int h = 0; h < 8; ++h)            // A: lane-lo K=0..7, lane-hi K=8..15
        a[h] = (_Float16)(diag[(size_t)nodec * WALK + 8 * hi + h] * amask);
    float lomask = (hi == 0) ? bmask : 0.f; // B: lane-lo K=0..15, lane-hi zero pad
    #pragma unroll
    for (int h = 0; h < 16; ++h)
        bb[h] = (_Float16)(W[(size_t)nc * WALK + h] * lomask);
    acc = __builtin_amdgcn_wmma_f32_16x16x32_f16(false, a, false, bb,
                                                 (short)0, acc, false, false);
#endif

    float bias = b[nc];
    if (n < ODIM) {
        #pragma unroll
        for (int v = 0; v < 8; ++v) {      // D: M = v + 8*hi, N = lane&15
            int rowi = base + v + 8 * hi;
            if (rowi < N) out[rowi * ODIM + n] = acc[v] + bias;
        }
    }
}

// ---------------- host orchestration ----------------

extern "C" void kernel_launch(void* const* d_in, const int* in_sizes, int n_in,
                              void* d_out, int out_size, void* d_ws, size_t ws_size,
                              hipStream_t stream) {
    (void)n_in; (void)ws_size;
    const int*   edge_index = (const int*)d_in[0];     // [2, E]
    const float* rw         = (const float*)d_in[2];   // [E]
    const float* W          = (const float*)d_in[3];   // [ODIM, WALK]
    const float* bias       = (const float*)d_in[4];   // [ODIM]
    float*       out        = (float*)d_out;

    const int E = in_sizes[0] / 2;
    const int N = out_size / ODIM;
    const int* row = edge_index;
    const int* col = edge_index + E;

    // carve workspace (256B aligned)
    char*  ws = (char*)d_ws;
    size_t o  = 0;
    auto carve = [&](size_t bytes) -> void* {
        o = (o + 255) & ~(size_t)255;
        void* p = ws + o;
        o += bytes;
        return p;
    };
    int*   cnt    = (int*)  carve((size_t)N * 4);
    int*   off    = (int*)  carve((size_t)(N + 1) * 4);
    int*   rows_s = (int*)  carve((size_t)E * 4);
    float* vals_s = (float*)carve((size_t)E * 4);
    float* diag   = (float*)carve((size_t)N * WALK * 4);
    float* bufA   = (float*)carve((size_t)N * CCHUNK * 4);
    float* bufB   = (float*)carve((size_t)N * CCHUNK * 4);

    // CSC build
    rwse_zero_i32 <<<(N + 255) / 256, 256, 0, stream>>>(cnt, N);
    rwse_count_in <<<(E + 255) / 256, 256, 0, stream>>>(col, E, cnt);
    rwse_scan_off <<<1, 1, 0, stream>>>(cnt, off, N);
    rwse_fill_csc <<<(N + 255) / 256, 256, 0, stream>>>(row, col, rw, E, off, rows_s, vals_s, N);

    // chunked diag((P^T)^k) for k=1..WALK
    const int nchunks = (N + CCHUNK - 1) / CCHUNK;
    for (int ci = 0; ci < nchunks; ++ci) {
        int start = ci * CCHUNK;
        rwse_init_onehot<<<N, TPB, 0, stream>>>(bufA, start);
        float* cur = bufA;
        float* nxt = bufB;
        for (int k = 0; k < WALK; ++k) {
            rwse_spmm_step<<<N, TPB, 0, stream>>>(off, rows_s, vals_s, cur, nxt, diag, start, k);
            float* tmp = cur; cur = nxt; nxt = tmp;
        }
    }

    // out = diag @ W^T + b  (WMMA, one wave per 16-node tile)
    rwse_linear_wmma<<<(N + 15) / 16, 32, 0, stream>>>(diag, W, bias, out, N);
}
